// SelfAttention_56032143344081
// MI455X (gfx1250) — matrix operand order repaired and checked by reading, not compiled, not run on hardware
//
#include <hip/hip_runtime.h>
#include <hip/hip_bf16.h>
#include <math.h>

// Problem constants (from reference): B=4, D=1024, L=2048, H=8, HD=128
#define B_  4
#define D_  1024
#define L_  2048
#define H_  8
#define HD_ 128

typedef __attribute__((ext_vector_type(16))) __bf16 v16bf;
typedef __attribute__((ext_vector_type(8)))  __bf16 v8bf;
typedef __attribute__((ext_vector_type(4)))  __bf16 v4bf;
typedef __attribute__((ext_vector_type(8)))  float  v8f;
typedef __attribute__((ext_vector_type(4)))  float  v4f;

static __device__ inline v16bf frag_join(v8bf lo, v8bf hi) {
  v16bf r;
#pragma unroll
  for (int i = 0; i < 8; ++i) { r[i] = lo[i]; r[i + 8] = hi[i]; }
  return r;
}

static __device__ inline v8f wmma_bf16(v16bf a, v16bf b, v8f c) {
  // D = A(16x32) * B(32x16) + C, f32 accumulate
  return __builtin_amdgcn_wmma_f32_16x16x32_bf16(false, a, false, b,
                                                 (short)0, c, false, false);
}

// ---------------------------------------------------------------------------
// Kernel 1: fused projections.  Unified output channel o in [0, 3D):
//   o in [0,D)   -> Q = Wq rows,  scaled by HD^-0.5, stored [B,H,L,HD] bf16
//   o in [D,2D)  -> K = W1 rows 0..D-1,              stored [B,H,L,HD] bf16
//   o in [2D,3D) -> V = W1 rows D..2D-1, stored TRANSPOSED [B,H,HD,L] bf16
// C[l,o] = sum_d query[b,d,l] * W[o,d].  Workgroup tile: 128 l x 128 o,
// 8 waves of 32x64.  K staged 32 deep; next slab is loaded into registers
// while the current slab's WMMAs run (software pipeline, loads overlapped).
// ---------------------------------------------------------------------------
__global__ __launch_bounds__(256) void proj_kernel(
    const float* __restrict__ query, const float* __restrict__ W1,
    const float* __restrict__ Wq,
    __bf16* __restrict__ Qbf, __bf16* __restrict__ Kbf,
    __bf16* __restrict__ Vt)
{
  __shared__ __align__(16) __bf16 As[128][40];  // [l][d-chunk], padded rows
  __shared__ __align__(16) __bf16 Ws[128][40];  // [o][d-chunk]

  const int tid  = threadIdx.x;
  const int lane = tid & 31;
  const int wave = tid >> 5;        // 0..7
  const int r    = lane & 15;
  const int g    = (lane >> 4) & 1;
  const int wm   = wave & 3;        // 0..3  (32-row block)
  const int wn   = wave >> 2;       // 0..1  (64-col block)

  const int ox   = blockIdx.x;      // 0..23
  const int type = ox >> 3;         // 0=Q, 1=K, 2=V
  const int o0   = (ox & 7) * 128;  // channel base within D
  const int l0   = blockIdx.y * 128;
  const int b    = blockIdx.z;

  const float* wbase = (type == 0) ? Wq : W1;
  const int    row0  = o0 + ((type == 2) ? D_ : 0);
  const float* qb    = query + (size_t)b * D_ * L_;

  v8f acc[2][4];
#pragma unroll
  for (int a = 0; a < 2; ++a)
#pragma unroll
    for (int c = 0; c < 4; ++c)
#pragma unroll
      for (int i = 0; i < 8; ++i) acc[a][c][i] = 0.0f;

  v4f qreg[4], wreg[4];
  auto load_slab = [&](int d0) {
#pragma unroll
    for (int rep = 0; rep < 4; ++rep) {
      int idx = tid + rep * 256;            // 0..1023 float4-groups
      int li  = (idx & 31) * 4;             // 0..124
      int kd  = idx >> 5;                   // 0..31
      qreg[rep] = *(const v4f*)&qb[(size_t)(d0 + kd) * L_ + (l0 + li)];
    }
#pragma unroll
    for (int rep = 0; rep < 4; ++rep) {
      int idx = tid + rep * 256;
      int kd4 = (idx & 7) * 4;              // 0..28
      int oi  = idx >> 3;                   // 0..127
      wreg[rep] = *(const v4f*)&wbase[(size_t)(row0 + oi) * D_ + (d0 + kd4)];
    }
  };
  auto store_slab = [&]() {
#pragma unroll
    for (int rep = 0; rep < 4; ++rep) {
      int idx = tid + rep * 256;
      int li  = (idx & 31) * 4;
      int kd  = idx >> 5;
#pragma unroll
      for (int j = 0; j < 4; ++j) As[li + j][kd] = (__bf16)qreg[rep][j];
    }
#pragma unroll
    for (int rep = 0; rep < 4; ++rep) {
      int idx = tid + rep * 256;
      int kd4 = (idx & 7) * 4;
      int oi  = idx >> 3;
      v4bf w4;
#pragma unroll
      for (int j = 0; j < 4; ++j) w4[j] = (__bf16)wreg[rep][j];
      *(v4bf*)&Ws[oi][kd4] = w4;
    }
  };

  load_slab(0);
  for (int d0 = 0; d0 < D_; d0 += 32) {
    store_slab();                 // waits on in-flight loads, fills LDS
    __syncthreads();              // slab visible to all waves
    if (d0 + 32 < D_) load_slab(d0 + 32);  // overlap next loads with WMMAs

    v16bf af[2], bfr[4];
#pragma unroll
    for (int a = 0; a < 2; ++a) {
      int row = wm * 32 + a * 16 + r;
      v8bf lo = *(const v8bf*)&As[row][g * 8];        // A: K = g*8..
      v8bf hi = *(const v8bf*)&As[row][16 + g * 8];   //    K = 16+g*8..
      af[a] = frag_join(lo, hi);
    }
#pragma unroll
    for (int c = 0; c < 4; ++c) {
      int row = wn * 64 + c * 16 + r;
      v8bf lo = *(const v8bf*)&Ws[row][g * 8];        // B: K = g*8..
      v8bf hi = *(const v8bf*)&Ws[row][16 + g * 8];   //    K = 16+g*8..
      bfr[c] = frag_join(lo, hi);
    }
#pragma unroll
    for (int a = 0; a < 2; ++a)
#pragma unroll
      for (int c = 0; c < 4; ++c)
        acc[a][c] = wmma_bf16(af[a], bfr[c], acc[a][c]);
    __syncthreads();              // all frag reads done before next overwrite
  }

  const float qscale = 0.08838834764831845f;  // 128^-0.5
#pragma unroll
  for (int a = 0; a < 2; ++a) {
#pragma unroll
    for (int c = 0; c < 4; ++c) {
      int o  = o0 + wn * 64 + c * 16 + r;             // column (N) = lane%16
      int h  = o >> 7;
      int hd = o & 127;
#pragma unroll
      for (int i = 0; i < 8; ++i) {
        int   ll = l0 + wm * 32 + a * 16 + i + 8 * g; // row (M) = i + 8g
        float v  = acc[a][c][i];
        if (type == 0)
          Qbf[(((size_t)b * H_ + h) * L_ + ll) * HD_ + hd] = (__bf16)(v * qscale);
        else if (type == 1)
          Kbf[(((size_t)b * H_ + h) * L_ + ll) * HD_ + hd] = (__bf16)v;
        else
          Vt[(((size_t)b * H_ + h) * HD_ + hd) * L_ + ll] = (__bf16)v;
      }
    }
  }
}

// ---------------------------------------------------------------------------
// Kernel 2: flash attention.  256 threads = 8 waves cover 128 query rows.
// 64 keys per iteration; K/V tiles are cooperatively staged through LDS,
// with the next tile's global loads overlapped with the current tile's
// WMMAs (register pipeline).  Online softmax amortized over 64 columns.
// ---------------------------------------------------------------------------
__global__ __launch_bounds__(256) void attn_kernel(
    const __bf16* __restrict__ Qbf, const __bf16* __restrict__ Kbf,
    const __bf16* __restrict__ Vt, const int* __restrict__ mask,
    float* __restrict__ out)
{
  __shared__ __align__(16) __bf16 Ks[64][136];   // 64 keys x 128 hd (padded)
  __shared__ __align__(16) __bf16 Vs[128][72];   // 128 hd x 64 keys (padded)
  __shared__ __align__(16) __bf16 Ps[8][16][72]; // per-wave P relayout

  const int tid  = threadIdx.x;
  const int lane = tid & 31;
  const int wave = tid >> 5;       // 0..7
  const int r    = lane & 15;
  const int g    = (lane >> 4) & 1;

  const int h  = blockIdx.y;
  const int b  = blockIdx.z;
  const int q0 = blockIdx.x * 128 + wave * 16;

  const size_t bh = (size_t)b * H_ + h;
  const __bf16* Qp = Qbf + (bh * L_ + q0) * HD_;
  const __bf16* Kp = Kbf + bh * L_ * HD_;
  const __bf16* Vp = Vt  + bh * HD_ * L_;
  const int*    mp = mask + (size_t)b * L_;

  // Q A-fragments for 4 K-chunks of 32 (HD = 128)
  v16bf qf[4];
#pragma unroll
  for (int c = 0; c < 4; ++c) {
    const __bf16* rowp = Qp + (size_t)r * HD_ + c * 32;
    v8bf lo = *(const v8bf*)(rowp + g * 8);
    v8bf hi = *(const v8bf*)(rowp + 16 + g * 8);
    qf[c] = frag_join(lo, hi);
  }

  v8f   o_acc[8];                  // 16 x 128 f32 accumulator
  float mrow[8], lrow[8];
#pragma unroll
  for (int t = 0; t < 8; ++t)
#pragma unroll
    for (int i = 0; i < 8; ++i) o_acc[t][i] = 0.0f;
#pragma unroll
  for (int i = 0; i < 8; ++i) { mrow[i] = -1e30f; lrow[i] = 0.0f; }

  v8bf kreg[4], vreg[4];
  int  mnext[4], mcur[4];
  auto load_tile = [&](int k0) {
#pragma unroll
    for (int rep = 0; rep < 4; ++rep) {
      int idx = tid + rep * 256;            // 0..1023 16B-groups
      int kk  = idx >> 4;                   // 0..63
      int off = (idx & 15) << 3;            // 0..120
      kreg[rep] = *(const v8bf*)(Kp + (size_t)(k0 + kk) * HD_ + off);
    }
#pragma unroll
    for (int rep = 0; rep < 4; ++rep) {
      int idx = tid + rep * 256;
      int hd  = idx >> 3;                   // 0..127
      int off = (idx & 7) << 3;             // 0..56
      vreg[rep] = *(const v8bf*)(Vp + (size_t)hd * L_ + k0 + off);
    }
#pragma unroll
    for (int t = 0; t < 4; ++t) mnext[t] = mp[k0 + t * 16 + r];
  };
  auto store_tile = [&]() {
#pragma unroll
    for (int rep = 0; rep < 4; ++rep) {
      int idx = tid + rep * 256;
      int kk  = idx >> 4;
      int off = (idx & 15) << 3;
      *(v8bf*)&Ks[kk][off] = kreg[rep];
    }
#pragma unroll
    for (int rep = 0; rep < 4; ++rep) {
      int idx = tid + rep * 256;
      int hd  = idx >> 3;
      int off = (idx & 7) << 3;
      *(v8bf*)&Vs[hd][off] = vreg[rep];
    }
  };

  load_tile(0);
  for (int k0 = 0; k0 < L_; k0 += 64) {
    __syncthreads();              // previous tile fully consumed
    store_tile();                 // waits in-flight loads, fills LDS
#pragma unroll
    for (int t = 0; t < 4; ++t) mcur[t] = mnext[t];
    __syncthreads();              // tile visible to all waves
    if (k0 + 64 < L_) load_tile(k0 + 64);  // overlap with compute

    // ---- scores: four 16x16 tiles over 64 keys ----
    v8f s[4];
#pragma unroll
    for (int t = 0; t < 4; ++t) {
#pragma unroll
      for (int i = 0; i < 8; ++i) s[t][i] = 0.0f;
#pragma unroll
      for (int c = 0; c < 4; ++c) {
        v8bf lo = *(const v8bf*)&Ks[t * 16 + r][c * 32 + g * 8];
        v8bf hi = *(const v8bf*)&Ks[t * 16 + r][c * 32 + 16 + g * 8];
        s[t] = wmma_bf16(qf[c], frag_join(lo, hi), s[t]);
      }
      float bias = mcur[t] ? 0.0f : -1e30f;  // key column n = lane%16
#pragma unroll
      for (int i = 0; i < 8; ++i) s[t][i] += bias;
    }

    // ---- online softmax (row m = i + 8g; reduce over 16-lane half) ----
    float f[8];
#pragma unroll
    for (int i = 0; i < 8; ++i) {
      float v = fmaxf(fmaxf(s[0][i], s[1][i]), fmaxf(s[2][i], s[3][i]));
#pragma unroll
      for (int off = 8; off >= 1; off >>= 1)
        v = fmaxf(v, __shfl_xor(v, off, 32));
      float mnew = fmaxf(mrow[i], v);
      f[i] = __expf(mrow[i] - mnew);
#pragma unroll
      for (int t = 0; t < 4; ++t) s[t][i] = __expf(s[t][i] - mnew);
      float sum = (s[0][i] + s[1][i]) + (s[2][i] + s[3][i]);
#pragma unroll
      for (int off = 8; off >= 1; off >>= 1)
        sum += __shfl_xor(sum, off, 32);
      lrow[i] = lrow[i] * f[i] + sum;
      mrow[i] = mnew;
    }
#pragma unroll
    for (int t = 0; t < 8; ++t)
#pragma unroll
      for (int i = 0; i < 8; ++i) o_acc[t][i] *= f[i];

    // ---- P: C-layout -> A-fragments via per-wave LDS (DS in-order) ----
#pragma unroll
    for (int i = 0; i < 8; ++i) {
      int m = i + 8 * g;
#pragma unroll
      for (int t = 0; t < 4; ++t)
        Ps[wave][m][t * 16 + r] = (__bf16)s[t][i];
    }
    v16bf pf[2];
#pragma unroll
    for (int u = 0; u < 2; ++u) {
      v8bf plo = *(const v8bf*)&Ps[wave][r][u * 32 + g * 8];
      v8bf phi = *(const v8bf*)&Ps[wave][r][u * 32 + 16 + g * 8];
      pf[u] = frag_join(plo, phi);
    }

    // ---- O += P(16x64) * V(64x128) from staged Vt tile ----
#pragma unroll
    for (int t = 0; t < 8; ++t) {
#pragma unroll
      for (int u = 0; u < 2; ++u) {
        v8bf lo = *(const v8bf*)&Vs[t * 16 + r][u * 32 + g * 8];
        v8bf hi = *(const v8bf*)&Vs[t * 16 + r][u * 32 + 16 + g * 8];
        o_acc[t] = wmma_bf16(pf[u], frag_join(lo, hi), o_acc[t]);
      }
    }
  }

  // ---- normalize and store to out[B, D, L] ----
  float inv[8];
#pragma unroll
  for (int i = 0; i < 8; ++i) inv[i] = 1.0f / lrow[i];
  float* ob = out + ((size_t)b * D_ + (size_t)h * HD_) * L_;
#pragma unroll
  for (int t = 0; t < 8; ++t) {
    int hd = t * 16 + r;
#pragma unroll
    for (int i = 0; i < 8; ++i) {
      int l = q0 + i + 8 * g;
      ob[(size_t)hd * L_ + l] = o_acc[t][i] * inv[i];
    }
  }
}

// ---------------------------------------------------------------------------
extern "C" void kernel_launch(void* const* d_in, const int* in_sizes, int n_in,
                              void* d_out, int out_size, void* d_ws, size_t ws_size,
                              hipStream_t stream) {
  const float* query = (const float*)d_in[0];  // [B, D, L] f32
  const int*   mask  = (const int*)d_in[1];    // [B, L] i32
  const float* W1    = (const float*)d_in[2];  // [2D, D] f32
  const float* Wq    = (const float*)d_in[3];  // [D, D] f32
  float*       out   = (float*)d_out;          // [B, D, L] f32

  // Workspace: Qbf, Kbf [B,H,L,HD] + Vt [B,H,HD,L], bf16 (~50.3 MB total)
  const size_t elems = (size_t)B_ * H_ * L_ * HD_;
  __bf16* Qbf = (__bf16*)d_ws;
  __bf16* Kbf = Qbf + elems;
  __bf16* Vt  = Kbf + elems;

  dim3 g1(24, L_ / 128, B_), b1(256);
  proj_kernel<<<g1, b1, 0, stream>>>(query, W1, Wq, Qbf, Kbf, Vt);

  dim3 g2(L_ / 128, H_, B_), b2(256);
  attn_kernel<<<g2, b2, 0, stream>>>(Qbf, Kbf, Vt, mask, out);
}
